// GridToBEV_26259430048548
// MI455X (gfx1250) — matrix operand order repaired and checked by reading, not compiled, not run on hardware
//
#include <hip/hip_runtime.h>
#include <hip/hip_bf16.h>
#include <math.h>

typedef __bf16 bf16_t;
typedef __attribute__((ext_vector_type(16))) __bf16 v16bf;
typedef __attribute__((ext_vector_type(8)))  __bf16 v8bf;
typedef __attribute__((ext_vector_type(8)))  float  v8f;
typedef __attribute__((ext_vector_type(4)))  unsigned int u32x4;
typedef __attribute__((ext_vector_type(8)))  int i32x8;
typedef __attribute__((ext_vector_type(4)))  int i32x4;

#define BB   4
#define MM   30000
#define NVX  300000
#define PP   32
#define CIN  128
#define CBEV 256
#define HY   180
#define WX   180
#define HW   (HY*WX)
#define OUTN (BB*CIN*HW)

#define BEVS 264   // padded bf16 stride for 256-wide LDS tile (TDM pads 4 dwords / 128 dwords)
#define KS   136   // padded bf16 stride for 128-wide LDS tile (TDM pads 4 dwords / 64 dwords)
#define FS   132   // padded f32 stride for 128-wide LDS tile

#define VSS (150.4f/1440.0f)
#define VSN (150.4f/180.0f)
#define ATT_SCALE 0.17677669529663689f  // 1/sqrt(32)

// branchless exact-ish GELU: Abramowitz-Stegun 7.1.26 erf (max err 1.5e-7)
__device__ __forceinline__ float geluf(float x){
  float z  = x * 0.70710678118654752f;
  float az = fabsf(z);
  float t  = 1.0f / (1.0f + 0.3275911f * az);
  float p  = ((((1.061405429f*t - 1.453152027f)*t + 1.421413741f)*t
              - 0.284496736f)*t + 0.254829592f)*t;
  float er = 1.0f - p * __expf(-az*az);
  er = copysignf(er, z);
  return 0.5f * x * (1.0f + er);
}
__device__ __forceinline__ bf16_t f2bf(float f){
  union { float f; unsigned u; } a; a.f = f;
  unsigned r = a.u + 0x7FFFu + ((a.u >> 16) & 1u);
  union { unsigned short s; bf16_t b; } o; o.s = (unsigned short)(r >> 16);
  return o.b;
}
__device__ __forceinline__ float bf2f(bf16_t b){
  union { bf16_t b; unsigned short s; } i; i.b = b;
  union { unsigned u; float f; } o; o.u = ((unsigned)i.s) << 16;
  return o.f;
}

// ---- Tensor Data Mover: async 2D bf16 tile load Global->LDS with row padding.
// pad_interval code: interval = (1<<code) 8-byte units; pad_amount code: (code+1) dwords
__device__ __forceinline__ void tdm_load_2d(const void* gptr, void* lptr,
    unsigned tensor_d0, unsigned tensor_d1, unsigned tile_d0, unsigned tile_d1,
    unsigned stride0, unsigned pad_interval, unsigned pad_amount)
{
  unsigned long long ga = (unsigned long long)gptr;
  unsigned la = (unsigned)(unsigned long long)lptr;
  u32x4 g0;
  g0[0] = 1u;                                       // count=1, user mode
  g0[1] = la;                                       // lds_addr
  g0[2] = (unsigned)(ga & 0xFFFFFFFFu);             // global_addr[31:0]
  g0[3] = (unsigned)((ga >> 32) & 0x1FFFFFFu) | (2u << 30); // addr[56:32] | type=2
  i32x8 g1;
  g1[0] = (int)((1u<<16) | (1u<<20) | (pad_interval<<22) | (pad_amount<<25)); // 2B elts, pad_enable
  g1[1] = (int)((tensor_d0 & 0xFFFFu) << 16);
  g1[2] = (int)((tensor_d0 >> 16) | ((tensor_d1 & 0xFFFFu) << 16));
  g1[3] = (int)((tensor_d1 >> 16) | (tile_d0 << 16));
  g1[4] = (int)tile_d1;                             // tile_dim2 = 0
  g1[5] = (int)stride0;                             // tensor_dim0_stride low
  g1[6] = 0; g1[7] = 0;
  i32x4 z4 = {0,0,0,0};
  i32x8 z8 = {0,0,0,0,0,0,0,0};
  __builtin_amdgcn_tensor_load_to_lds(g0, g1, z4, z4, z8, 0);
}

// A fragment (16x32 bf16): lane l -> row0+(l&15), K = k0+(l>>4)*8 +[0..7] and +16..23
__device__ __forceinline__ v16bf load_a_frag(const bf16_t* base, int ld, int row0, int k0){
  const int l = threadIdx.x & 31;
  const bf16_t* p = base + (row0 + (l & 15)) * ld + k0 + ((l >> 4) << 3);
  union { v16bf v; v8bf h[2]; } u;
  u.h[0] = *(const v8bf*)(p);
  u.h[1] = *(const v8bf*)(p + 16);
  return u.v;
}
// B fragment (32x16 bf16) from (out_c, k) row-major: lane l -> col0+(l&15), K = k0+(l>>4)*8 +[0..7] and +16..23
__device__ __forceinline__ v16bf load_b_frag(const bf16_t* base, int ld, int col0, int k0){
  const int l = threadIdx.x & 31;
  const bf16_t* p = base + (col0 + (l & 15)) * ld + k0 + ((l >> 4) << 3);
  union { v16bf v; v8bf h[2]; } u;
  u.h[0] = *(const v8bf*)(p);
  u.h[1] = *(const v8bf*)(p + 16);
  return u.v;
}
#define WMMA_BF16(a,b,c) __builtin_amdgcn_wmma_f32_16x16x32_bf16(false,(a),false,(b),(short)0,(c),false,false)
#define V8F_ZERO {0.f,0.f,0.f,0.f,0.f,0.f,0.f,0.f}

// -------------------------------------------------------------------------
__global__ void k_zero(float* __restrict__ out, int n){
  int i = blockIdx.x * 256 + threadIdx.x;
  if (i < n) out[i] = 0.0f;
}
__global__ void k_cvt(const float* __restrict__ s, bf16_t* __restrict__ d, int n){
  int i = blockIdx.x * 256 + threadIdx.x;
  if (i < n) d[i] = f2bf(s[i]);
}

// -------------------------------------------------------------------------
// 16 queries / block. bev gather -> shortcut GEMM (K=256) -> gelu -> LN -> qh GEMM
__global__ __launch_bounds__(256) void k_shortcut_q(
    const float* __restrict__ bev, const int* __restrict__ nidx,
    const float* __restrict__ b_bev, const float* __restrict__ g_q, const float* __restrict__ be_q,
    const bf16_t* __restrict__ wbev_bf, const bf16_t* __restrict__ win_bf,
    const float* __restrict__ b_in,
    float* __restrict__ shortcut, float* __restrict__ qh)
{
  __shared__ __align__(16) bf16_t wbevS[128*BEVS];
  __shared__ __align__(16) bf16_t wqS[128*KS];
  __shared__ __align__(16) bf16_t bevt[16*BEVS];
  __shared__ __align__(16) float  shs[16*FS];
  __shared__ __align__(16) bf16_t qlnt[16*KS];
  __shared__ float mu[16], rs[16];

  const int m0 = blockIdx.x * 16;
  const int t = threadIdx.x;
  const int wave = t >> 5, l = t & 31, n = l & 15, half = l >> 4;

  if (t < 32){  // wave 0: async TDM stage of weights with row padding
    tdm_load_2d(wbev_bf, wbevS, CBEV, 128, CBEV, 128, CBEV, 6, 3); // 512B rows
    tdm_load_2d(win_bf,  wqS,   CIN,  128, CIN,  128, CIN,  5, 3); // 256B rows
  }

  { // gather bev tile while TDM runs
    int r = t >> 4, seg = t & 15;
    int m = m0 + r;
    int b = nidx[m*4+0], y = nidx[m*4+2], x = nidx[m*4+3];
    long base = (long)b * CBEV * HW + (long)y * WX + x;
    #pragma unroll
    for (int jj = 0; jj < 16; ++jj){
      int j = seg*16 + jj;
      bevt[r*BEVS + j] = f2bf(bev[base + (long)j * HW]);
    }
  }
  if (t < 32) __builtin_amdgcn_s_wait_tensorcnt(1);  // wbev done (in-order)
  __syncthreads();

  v8f acc = V8F_ZERO;
  #pragma unroll
  for (int kk = 0; kk < 8; ++kk){
    v16bf a = load_a_frag(bevt,  BEVS, 0,       kk*32);
    v16bf b = load_b_frag(wbevS, BEVS, wave*16, kk*32);
    acc = WMMA_BF16(a, b, acc);
  }
  #pragma unroll
  for (int r8 = 0; r8 < 8; ++r8){
    int row = r8 + 8*half;
    int c = wave*16 + n;
    float s = geluf(acc[r8] + b_bev[c]);
    shortcut[(long)(m0+row)*CIN + c] = s;
    shs[row*FS + c] = s;
  }
  __syncthreads();
  if (t < 16){
    float sum = 0.f, sq = 0.f;
    for (int c = 0; c < CIN; ++c){ float v = shs[t*FS+c]; sum += v; sq += v*v; }
    float m_ = sum * (1.f/CIN);
    mu[t] = m_; rs[t] = rsqrtf(sq*(1.f/CIN) - m_*m_ + 1e-5f);
  }
  __syncthreads();
  #pragma unroll
  for (int i = 0; i < 8; ++i){
    int idx = t*8 + i; int r = idx >> 7, c = idx & 127;
    qlnt[r*KS + c] = f2bf((shs[r*FS+c] - mu[r]) * rs[r] * g_q[c] + be_q[c]);
  }
  if (t < 32) __builtin_amdgcn_s_wait_tensorcnt(0);  // wq done
  __syncthreads();
  v8f acc2 = V8F_ZERO;
  #pragma unroll
  for (int kk = 0; kk < 4; ++kk){
    v16bf a = load_a_frag(qlnt, KS, 0,       kk*32);
    v16bf b = load_b_frag(wqS,  KS, wave*16, kk*32);
    acc2 = WMMA_BF16(a, b, acc2);
  }
  #pragma unroll
  for (int r8 = 0; r8 < 8; ++r8){
    int row = r8 + 8*half;
    int c = wave*16 + n;
    qh[(long)(m0+row)*CIN + c] = acc2[r8] + b_in[c];
  }
}

// -------------------------------------------------------------------------
// 8 queries / block (256 key rows). gather+pos+LN -> LDS; kh WMMA; logits/softmax;
// vh WMMA (reusing kh buffer + TDM-restaged wv); attn*v -> av (bf16, ws)
__global__ __launch_bounds__(256) void k_attn(
    const float* __restrict__ vf, const int* __restrict__ spidx,
    const int* __restrict__ nidx, const int* __restrict__ kidx,
    const float* __restrict__ w_pos, const float* __restrict__ b_pos,
    const float* __restrict__ g_k, const float* __restrict__ be_k,
    const bf16_t* __restrict__ win_bf, const float* __restrict__ b_in,
    const float* __restrict__ qh, bf16_t* __restrict__ av_out)
{
  __shared__ __align__(16) bf16_t klnS[256*KS];
  __shared__ __align__(16) bf16_t khvS[256*KS];
  __shared__ __align__(16) bf16_t wS[128*KS];     // wk, then wv (TDM restage)
  __shared__ __align__(16) float  qhS[8*CIN];
  __shared__ __align__(16) float  logitsS[32*32];
  __shared__ unsigned char maskS[256];

  const int m0 = blockIdx.x * 8;
  const int t = threadIdx.x;
  const int wave = t >> 5, l = t & 31, n = l & 15, half = l >> 4;

  if (t < 32)  // stage wk while we gather/normalize keys
    tdm_load_2d(win_bf + CIN*CIN, wS, CIN, 128, CIN, 128, CIN, 5, 3);

  // stage qh tile
  for (int i = t; i < 8*CIN; i += 256)
    qhS[i] = qh[(long)(m0 + (i >> 7))*CIN + (i & 127)];

  // stage A: wave w owns query m0+w; one key row per iteration, lane -> 4 channels
  {
    int m = m0 + wave;
    float qx = ((float)nidx[m*4+3] + 0.5f) * VSN - 75.2f;
    float qy = ((float)nidx[m*4+2] + 0.5f) * VSN - 75.2f;
    float qz = 1.0f;
    float wpx[4], wpy[4], wpz[4], bp[4], gkv[4], bkv[4];
    #pragma unroll
    for (int j = 0; j < 4; ++j){
      int c = l*4 + j;
      wpx[j] = w_pos[c*3+0]; wpy[j] = w_pos[c*3+1]; wpz[j] = w_pos[c*3+2];
      bp[j] = b_pos[c]; gkv[j] = g_k[c]; bkv[j] = be_k[c];
    }
    const int* krow = kidx + (long)m*PP;
    for (int p = 0; p < PP; ++p){
      int row = wave*PP + p;
      int ki = krow[p];
      int safe = ki < 0 ? 0 : ki;
      if (p + 1 < PP){             // prefetch next key's feature row
        int kin = krow[p+1]; int sn = kin < 0 ? 0 : kin;
        __builtin_prefetch(vf + (long)sn*CIN + l*4, 0, 0);
      }
      float dx = ((float)spidx[safe*4+3] + 0.5f) * VSS - 75.2f - qx;
      float dy = ((float)spidx[safe*4+2] + 0.5f) * VSS - 75.2f - qy;
      float dz = ((float)spidx[safe*4+1] + 0.5f) * 0.15f - 2.0f - qz;
      float4 f = *(const float4*)(vf + (long)safe*CIN + l*4);
      float val[4] = {f.x, f.y, f.z, f.w};
      float s = 0.f, sq = 0.f;
      #pragma unroll
      for (int j = 0; j < 4; ++j){
        val[j] += geluf(wpx[j]*dx + wpy[j]*dy + wpz[j]*dz + bp[j]);
        s += val[j]; sq += val[j]*val[j];
      }
      #pragma unroll
      for (int off = 16; off > 0; off >>= 1){
        s  += __shfl_xor(s,  off);
        sq += __shfl_xor(sq, off);
      }
      float mean = s * (1.f/CIN);
      float rstd = rsqrtf(sq*(1.f/CIN) - mean*mean + 1e-5f);
      union { bf16_t b[4]; uint2 u; } pk;
      #pragma unroll
      for (int j = 0; j < 4; ++j)
        pk.b[j] = f2bf((val[j] - mean) * rstd * gkv[j] + bkv[j]);
      *(uint2*)&klnS[row*KS + l*4] = pk.u;
      if (l == 0) maskS[row] = (ki < 0);
    }
  }
  if (t < 32) __builtin_amdgcn_s_wait_tensorcnt(0);
  __syncthreads();

  // kh = kln @ wk^T + bk  -> khvS (bf16)
  for (int mt = 0; mt < 16; ++mt){
    v8f acc = V8F_ZERO;
    #pragma unroll
    for (int kk = 0; kk < 4; ++kk){
      v16bf a = load_a_frag(klnS, KS, mt*16,   kk*32);
      v16bf b = load_b_frag(wS,   KS, wave*16, kk*32);
      acc = WMMA_BF16(a, b, acc);
    }
    #pragma unroll
    for (int r8 = 0; r8 < 8; ++r8){
      int row = mt*16 + r8 + 8*half;
      int c = wave*16 + n;
      khvS[row*KS + c] = f2bf(acc[r8] + b_in[CIN + c]);
    }
  }
  __syncthreads();

  if (t < 32)  // restage wS with wv; overlaps with logits/softmax below
    tdm_load_2d(win_bf + 2*CIN*CIN, wS, CIN, 128, CIN, 128, CIN, 5, 3);

  // logits: 32 (query,head) pairs, 8 threads each, 4 keys per thread
  {
    int pair = t >> 3, sub = t & 7;
    int ml = pair >> 2, h = pair & 3;
    #pragma unroll
    for (int pp = 0; pp < 4; ++pp){
      int p = sub*4 + pp;
      const bf16_t* kr = &khvS[(ml*PP + p)*KS + h*32];
      const float*  qr = &qhS[ml*CIN + h*32];
      float a = 0.f;
      #pragma unroll
      for (int d = 0; d < 32; ++d) a += qr[d] * bf2f(kr[d]);
      a *= ATT_SCALE;
      if (maskS[ml*PP + p]) a = -__builtin_inff();
      logitsS[pair*32 + p] = a;
    }
  }
  __syncthreads();
  if (t < 32){
    float mx = -__builtin_inff();
    for (int p = 0; p < 32; ++p) mx = fmaxf(mx, logitsS[t*32+p]);
    float sum = 0.f;
    for (int p = 0; p < 32; ++p){ float e = __expf(logitsS[t*32+p] - mx); logitsS[t*32+p] = e; sum += e; }
    float inv = 1.f / sum;
    for (int p = 0; p < 32; ++p) logitsS[t*32+p] *= inv;
    __builtin_amdgcn_s_wait_tensorcnt(0);   // wv staged
  }
  __syncthreads();

  // vh = kln @ wv^T + bv  -> khvS (overwrite)
  for (int mt = 0; mt < 16; ++mt){
    v8f acc = V8F_ZERO;
    #pragma unroll
    for (int kk = 0; kk < 4; ++kk){
      v16bf a = load_a_frag(klnS, KS, mt*16,   kk*32);
      v16bf b = load_b_frag(wS,   KS, wave*16, kk*32);
      acc = WMMA_BF16(a, b, acc);
    }
    #pragma unroll
    for (int r8 = 0; r8 < 8; ++r8){
      int row = mt*16 + r8 + 8*half;
      int c = wave*16 + n;
      khvS[row*KS + c] = f2bf(acc[r8] + b_in[2*CIN + c]);
    }
  }
  __syncthreads();

  // av[m,c] = sum_p attn[m,h(c),p] * vh[m,p,c]
  #pragma unroll
  for (int i = 0; i < 4; ++i){
    int o = t*4 + i;
    int ml = o >> 7, c = o & 127, h = c >> 5;
    const float* arow = &logitsS[(ml*4 + h)*32];
    float a = 0.f;
    #pragma unroll
    for (int p = 0; p < 32; ++p) a += arow[p] * bf2f(khvS[(ml*PP + p)*KS + c]);
    av_out[(long)(m0+ml)*CIN + c] = f2bf(a);
  }
}

// -------------------------------------------------------------------------
// 16 queries / block: out-proj + residual, LN, fc1+gelu, fc2 + residual, scatter
__global__ __launch_bounds__(256) void k_mlp_scatter(
    const bf16_t* __restrict__ av, const float* __restrict__ shortcut,
    const bf16_t* __restrict__ wout_bf, const float* __restrict__ b_out,
    const float* __restrict__ g_f, const float* __restrict__ be_f,
    const bf16_t* __restrict__ wfc1_bf, const float* __restrict__ b_fc1,
    const bf16_t* __restrict__ wfc2_bf, const float* __restrict__ b_fc2,
    const int* __restrict__ nidx, float* __restrict__ out)
{
  __shared__ __align__(16) bf16_t woutS[128*KS];
  __shared__ __align__(16) bf16_t wfc1S[128*KS];
  __shared__ __align__(16) bf16_t wfc2S[128*KS];
  __shared__ __align__(16) bf16_t avt[16*KS];
  __shared__ __align__(16) float  xfS[16*FS];
  __shared__ __align__(16) bf16_t ylnt[16*KS];
  __shared__ __align__(16) bf16_t h1t[16*KS];
  __shared__ float mu[16], rs[16];

  const int m0 = blockIdx.x * 16;
  const int t = threadIdx.x;
  const int wave = t >> 5, l = t & 31, n = l & 15, half = l >> 4;

  if (t < 32){  // async stage all three weights; waits staggered (in-order completion)
    tdm_load_2d(wout_bf, woutS, CIN, 128, CIN, 128, CIN, 5, 3);
    tdm_load_2d(wfc1_bf, wfc1S, CIN, 128, CIN, 128, CIN, 5, 3);
    tdm_load_2d(wfc2_bf, wfc2S, CIN, 128, CIN, 128, CIN, 5, 3);
  }

  #pragma unroll
  for (int i = 0; i < 8; ++i){
    int idx = t*8 + i; int r = idx >> 7, c = idx & 127;
    avt[r*KS + c] = av[(long)(m0+r)*CIN + c];
  }
  if (t < 32) __builtin_amdgcn_s_wait_tensorcnt(2);  // wout ready
  __syncthreads();

  v8f acc = V8F_ZERO;
  #pragma unroll
  for (int kk = 0; kk < 4; ++kk){
    v16bf a = load_a_frag(avt,   KS, 0,       kk*32);
    v16bf b = load_b_frag(woutS, KS, wave*16, kk*32);
    acc = WMMA_BF16(a, b, acc);
  }
  #pragma unroll
  for (int r8 = 0; r8 < 8; ++r8){
    int row = r8 + 8*half;
    int c = wave*16 + n;
    xfS[row*FS + c] = acc[r8] + b_out[c] + shortcut[(long)(m0+row)*CIN + c];
  }
  __syncthreads();
  if (t < 16){
    float sum = 0.f, sq = 0.f;
    for (int c = 0; c < CIN; ++c){ float v = xfS[t*FS+c]; sum += v; sq += v*v; }
    float m_ = sum * (1.f/CIN);
    mu[t] = m_; rs[t] = rsqrtf(sq*(1.f/CIN) - m_*m_ + 1e-5f);
  }
  __syncthreads();
  #pragma unroll
  for (int i = 0; i < 8; ++i){
    int idx = t*8 + i; int r = idx >> 7, c = idx & 127;
    ylnt[r*KS + c] = f2bf((xfS[r*FS+c] - mu[r]) * rs[r] * g_f[c] + be_f[c]);
  }
  if (t < 32) __builtin_amdgcn_s_wait_tensorcnt(1);  // wfc1 ready
  __syncthreads();

  v8f acc1 = V8F_ZERO;
  #pragma unroll
  for (int kk = 0; kk < 4; ++kk){
    v16bf a = load_a_frag(ylnt,  KS, 0,       kk*32);
    v16bf b = load_b_frag(wfc1S, KS, wave*16, kk*32);
    acc1 = WMMA_BF16(a, b, acc1);
  }
  #pragma unroll
  for (int r8 = 0; r8 < 8; ++r8){
    int row = r8 + 8*half;
    int c = wave*16 + n;
    h1t[row*KS + c] = f2bf(geluf(acc1[r8] + b_fc1[c]));
  }
  if (t < 32) __builtin_amdgcn_s_wait_tensorcnt(0);  // wfc2 ready
  __syncthreads();

  v8f acc2 = V8F_ZERO;
  #pragma unroll
  for (int kk = 0; kk < 4; ++kk){
    v16bf a = load_a_frag(h1t,   KS, 0,       kk*32);
    v16bf b = load_b_frag(wfc2S, KS, wave*16, kk*32);
    acc2 = WMMA_BF16(a, b, acc2);
  }
  #pragma unroll
  for (int r8 = 0; r8 < 8; ++r8){
    int row = r8 + 8*half;
    int c = wave*16 + n;
    int m = m0 + row;
    float feats = acc2[r8] + b_fc2[c] + xfS[row*FS + c];
    int b = nidx[m*4+0], y = nidx[m*4+2], x = nidx[m*4+3];
    out[((long)b*CIN + c)*HW + (long)y*WX + x] = feats;
  }
}

// -------------------------------------------------------------------------
extern "C" void kernel_launch(void* const* d_in, const int* in_sizes, int n_in,
                              void* d_out, int out_size, void* d_ws, size_t ws_size,
                              hipStream_t stream) {
  const float* vf    = (const float*)d_in[0];
  const int*   sp    = (const int*)  d_in[1];
  const int*   ni    = (const int*)  d_in[2];
  const int*   kix   = (const int*)  d_in[3];
  const float* bev   = (const float*)d_in[4];
  const float* w_pos = (const float*)d_in[5];
  const float* b_pos = (const float*)d_in[6];
  const float* w_bev = (const float*)d_in[7];
  const float* b_bev = (const float*)d_in[8];
  const float* g_q   = (const float*)d_in[9];
  const float* be_q  = (const float*)d_in[10];
  const float* g_k   = (const float*)d_in[11];
  const float* be_k  = (const float*)d_in[12];
  const float* g_f   = (const float*)d_in[13];
  const float* be_f  = (const float*)d_in[14];
  const float* w_in  = (const float*)d_in[15];
  const float* b_in  = (const float*)d_in[16];
  const float* w_out = (const float*)d_in[17];
  const float* b_out = (const float*)d_in[18];
  const float* w_fc1 = (const float*)d_in[19];
  const float* b_fc1 = (const float*)d_in[20];
  const float* w_fc2 = (const float*)d_in[21];
  const float* b_fc2 = (const float*)d_in[22];
  float* out = (float*)d_out;
  char* ws = (char*)d_ws;

  constexpr size_t WIN_OFF  = 0;                                  // 384x128 bf16
  constexpr size_t WBEV_OFF = WIN_OFF  + (size_t)384*128*2;
  constexpr size_t WOUT_OFF = WBEV_OFF + (size_t)128*256*2;
  constexpr size_t WFC1_OFF = WOUT_OFF + (size_t)128*128*2;
  constexpr size_t WFC2_OFF = WFC1_OFF + (size_t)128*128*2;
  constexpr size_t SC_OFF   = WFC2_OFF + (size_t)128*128*2;       // f32 M*128
  constexpr size_t QH_OFF   = SC_OFF   + (size_t)MM*CIN*4;        // f32 M*128
  constexpr size_t AV_OFF   = QH_OFF   + (size_t)MM*CIN*4;        // bf16 M*128

  bf16_t* win_bf  = (bf16_t*)(ws + WIN_OFF);
  bf16_t* wbev_bf = (bf16_t*)(ws + WBEV_OFF);
  bf16_t* wout_bf = (bf16_t*)(ws + WOUT_OFF);
  bf16_t* wfc1_bf = (bf16_t*)(ws + WFC1_OFF);
  bf16_t* wfc2_bf = (bf16_t*)(ws + WFC2_OFF);
  float*  shortcut = (float*)(ws + SC_OFF);
  float*  qh       = (float*)(ws + QH_OFF);
  bf16_t* av       = (bf16_t*)(ws + AV_OFF);

  k_zero<<<(OUTN + 255)/256, 256, 0, stream>>>(out, OUTN);
  k_cvt<<<(384*128 + 255)/256, 256, 0, stream>>>(w_in,  win_bf,  384*128);
  k_cvt<<<(128*256 + 255)/256, 256, 0, stream>>>(w_bev, wbev_bf, 128*256);
  k_cvt<<<(128*128 + 255)/256, 256, 0, stream>>>(w_out, wout_bf, 128*128);
  k_cvt<<<(128*128 + 255)/256, 256, 0, stream>>>(w_fc1, wfc1_bf, 128*128);
  k_cvt<<<(128*128 + 255)/256, 256, 0, stream>>>(w_fc2, wfc2_bf, 128*128);

  k_shortcut_q<<<MM/16, 256, 0, stream>>>(bev, ni, b_bev, g_q, be_q,
                                          wbev_bf, win_bf, b_in, shortcut, qh);
  k_attn<<<MM/8, 256, 0, stream>>>(vf, sp, ni, kix, w_pos, b_pos, g_k, be_k,
                                   win_bf, b_in, qh, av);
  k_mlp_scatter<<<MM/16, 256, 0, stream>>>(av, shortcut, wout_bf, b_out, g_f, be_f,
                                           wfc1_bf, b_fc1, wfc2_bf, b_fc2, ni, out);
}
